// Merge_MixtralSparseMoeBlock_24696061952354
// MI455X (gfx1250) — matrix operand order repaired and checked by reading, not compiled, not run on hardware
//
#include <hip/hip_runtime.h>
#include <hip/hip_bf16.h>
#include <cstdint>

// ---------------- problem constants ----------------
#define TOK   4096
#define HID   4096
#define INTER 14336
#define RNK   318
#define RP    320          // rank padded to multiple of 32

typedef __attribute__((ext_vector_type(16))) __bf16 v16bf;
typedef __attribute__((ext_vector_type(8)))  __bf16 v8bf;
typedef __attribute__((ext_vector_type(8)))  float  v8f;
typedef int v4i __attribute__((vector_size(16)));

static __device__ __forceinline__ __bf16 f2bf(float f) {
    unsigned u = __builtin_bit_cast(unsigned, f);
    unsigned r = u + 0x7FFFu + ((u >> 16) & 1u);   // round-to-nearest-even
    unsigned short h = (unsigned short)(r >> 16);
    return __builtin_bit_cast(__bf16, h);
}

// ---------------- CDNA5 async global->LDS path ----------------
#if defined(__HIP_DEVICE_COMPILE__) && __has_builtin(__builtin_amdgcn_global_load_async_to_lds_b128)
#define HAVE_ASYNC 1
#else
#define HAVE_ASYNC 0
#endif

#if HAVE_ASYNC
typedef __attribute__((address_space(1))) v4i as1_v4i;
typedef __attribute__((address_space(3))) v4i as3_v4i;
static __device__ __forceinline__ void async_cp16(const void* g, void* l) {
    __builtin_amdgcn_global_load_async_to_lds_b128(
        (as1_v4i*)(uintptr_t)g, (as3_v4i*)(uintptr_t)l, 0, 0);
}
#endif

template <int N>
static __device__ __forceinline__ void wait_async() {
#if defined(__HIP_DEVICE_COMPILE__)
#if __has_builtin(__builtin_amdgcn_s_wait_asynccnt)
    __builtin_amdgcn_s_wait_asynccnt(N);
#else
    asm volatile("s_wait_asynccnt %0" :: "n"(N) : "memory");
#endif
#endif
}

// ---------------- fp32 -> bf16 convert with zero padding ----------------
__global__ __launch_bounds__(256) void cvt_pad_kernel(const float* __restrict__ src,
                                                      __bf16* __restrict__ dst,
                                                      int sr, int sc, int dr, int dc) {
    int i = blockIdx.x * 256 + threadIdx.x;
    if (i >= dr * dc) return;
    int r = i / dc, c = i - r * dc;
    float v = (r < sr && c < sc) ? src[r * sc + c] : 0.0f;
    dst[i] = f2bf(v);
}

// ---------------- tile loaders (global -> LDS) ----------------
#define LDST  40   // bf16 LDS row stride (32 + 8 pad, keeps 16B alignment)
#define LDSTF 36   // f32  LDS row stride (32 + 4 pad, keeps 16B alignment)

template <int ROWS>     // bf16 tile: ROWS x 32 bf16
static __device__ __forceinline__ void load_tile_bf16(__bf16* lds, const __bf16* src,
                                                      int ld, int t) {
    constexpr int VECS = ROWS * 4;   // 16B vectors
#if HAVE_ASYNC
#pragma unroll
    for (int i = 0; i < VECS / 256; ++i) {
        int id = t + i * 256, row = id >> 2, co = (id & 3) * 8;
        async_cp16(src + (size_t)row * ld + co, lds + row * LDST + co);
    }
#else
    uint4 tmp[VECS / 256];
#pragma unroll
    for (int i = 0; i < VECS / 256; ++i) {
        int id = t + i * 256, row = id >> 2, co = (id & 3) * 8;
        tmp[i] = *(const uint4*)(src + (size_t)row * ld + co);
    }
#pragma unroll
    for (int i = 0; i < VECS / 256; ++i) {
        int id = t + i * 256, row = id >> 2, co = (id & 3) * 8;
        *(uint4*)(&lds[row * LDST + co]) = tmp[i];
    }
#endif
}

template <int ROWS>     // f32 tile: ROWS x 32 f32 (kept fp32 in LDS)
static __device__ __forceinline__ void load_tile_f32(float* lds, const float* src,
                                                     int ld, int t) {
    constexpr int VECS = ROWS * 8;   // 16B vectors
#if HAVE_ASYNC
#pragma unroll
    for (int i = 0; i < VECS / 256; ++i) {
        int id = t + i * 256, row = id >> 3, co = (id & 7) * 4;
        async_cp16(src + (size_t)row * ld + co, lds + row * LDSTF + co);
    }
#else
    float4 tmp[VECS / 256];
#pragma unroll
    for (int i = 0; i < VECS / 256; ++i) {
        int id = t + i * 256, row = id >> 3, co = (id & 7) * 4;
        tmp[i] = *(const float4*)(src + (size_t)row * ld + co);
    }
#pragma unroll
    for (int i = 0; i < VECS / 256; ++i) {
        int id = t + i * 256, row = id >> 3, co = (id & 7) * 4;
        *(float4*)(&lds[row * LDSTF + co]) = tmp[i];
    }
#endif
}

// ---------------- fragment gathers from LDS ----------------
static __device__ __forceinline__ v16bf load_frag(const __bf16* lds, int row, int o0, int o1) {
    v8bf lo = *(const v8bf*)(lds + row * LDST + o0);
    v8bf hi = *(const v8bf*)(lds + row * LDST + o1);
    return __builtin_shufflevector(lo, hi, 0, 1, 2, 3, 4, 5, 6, 7,
                                           8, 9, 10, 11, 12, 13, 14, 15);
}
// B fragment from an fp32 LDS tile, converting to bf16 in-register
static __device__ __forceinline__ v16bf load_frag_f32(const float* lds, int row, int kh) {
    v16bf out;
#pragma unroll
    for (int j = 0; j < 16; j += 4) {
        float4 v = *(const float4*)(lds + row * LDSTF + kh + j);
        out[j + 0] = f2bf(v.x); out[j + 1] = f2bf(v.y);
        out[j + 2] = f2bf(v.z); out[j + 3] = f2bf(v.w);
    }
    return out;
}

// ---------------- fused WMMA GEMM ----------------
// C[M,N] = A(bf16)[M,K] * B[N,K]^T  (+ lowrank A2[M,RP]*LB[N,RP]^T) (+ SiLU fuse)
// MODE 0: dual-B bf16 -> two bf16 (t1,t3)
// MODE 1: dual-B fp32 (W1,W3) + dual lowrank + silu(gate)*up -> bf16 z
// MODE 2: single-B bf16 (dv2) -> bf16 t2
// MODE 3: single-B fp32 (W2) + lowrank -> fp32 out
template <int MODE>
__global__ __launch_bounds__(256) void gemm_ker(
    const __bf16* __restrict__ A, int lda,
    const void* __restrict__ B1, const void* __restrict__ B2, int ldb,
    const __bf16* __restrict__ A2_1, const __bf16* __restrict__ A2_2,
    const __bf16* __restrict__ LB1, const __bf16* __restrict__ LB2,
    void* __restrict__ C1, void* __restrict__ C2, int ldc, int K) {

    constexpr bool DUAL = (MODE == 0 || MODE == 1);
    constexpr bool BF32 = (MODE == 1 || MODE == 3);
    constexpr bool LR   = (MODE == 1 || MODE == 3);
    constexpr int  NB   = DUAL ? 2 : 1;
    constexpr int  OPS  = 2 + NB * (BF32 ? 2 : 1);  // async b128 ops / thread / tile

    constexpr size_t ABUF = (size_t)128 * LDST * sizeof(__bf16);
    constexpr size_t BBUF = BF32 ? (size_t)64 * LDSTF * sizeof(float)
                                 : (size_t)64 * LDST * sizeof(__bf16);
    __shared__ __align__(16) char smem[2 * ABUF + 2 * NB * BBUF];
    // NOTE: compute LDS sub-buffer pointers on the fly; a constant-initialized
    // local pointer array becomes a static initializer with an addrspacecast,
    // which ld.lld rejects.
    auto sA = [&](int buf) -> __bf16* { return (__bf16*)(smem + (size_t)buf * ABUF); };
    auto sB = [&](int buf, int mat) -> char* {
        return smem + 2 * ABUF + ((size_t)buf * NB + mat) * BBUF;
    };

    const int t    = threadIdx.x;
    const int m0   = blockIdx.y * 128;
    const int n0   = blockIdx.x * 64;
    const int wave = t >> 5;
    const int lane = t & 31;
    const int wm   = wave & 3;     // 4 waves x 32 rows
    const int wn   = wave >> 2;    // 2 waves x 32 cols
    const int h    = lane >> 4;
    const int r    = lane & 15;

    v8f acc1[2][2] = {};
    v8f acc2[2][2] = {};

    const __bf16* Bb1 = (const __bf16*)B1;
    const __bf16* Bb2 = (const __bf16*)B2;
    const float*  Bf1 = (const float*)B1;
    const float*  Bf2 = (const float*)B2;

    auto issue_main = [&](int buf, int kb) {
        load_tile_bf16<128>(sA(buf), A + (size_t)m0 * lda + kb, lda, t);
        if constexpr (BF32) {
            load_tile_f32<64>((float*)sB(buf, 0), Bf1 + (size_t)n0 * ldb + kb, ldb, t);
            if constexpr (DUAL)
                load_tile_f32<64>((float*)sB(buf, 1), Bf2 + (size_t)n0 * ldb + kb, ldb, t);
        } else {
            load_tile_bf16<64>((__bf16*)sB(buf, 0), Bb1 + (size_t)n0 * ldb + kb, ldb, t);
            if constexpr (DUAL)
                load_tile_bf16<64>((__bf16*)sB(buf, 1), Bb2 + (size_t)n0 * ldb + kb, ldb, t);
        }
    };

    // ---------- main K loop: double-buffered async pipeline ----------
    const int nk = K / 32;
    issue_main(0, 0);
    for (int it = 0; it < nk; ++it) {
        const int cur = it & 1;
        __syncthreads();                       // readers of buf cur^1 are done
        if (it + 1 < nk) { issue_main(cur ^ 1, (it + 1) * 32); wait_async<OPS>(); }
        else             { wait_async<0>(); }
        __syncthreads();                       // tile `it` visible to all waves

        v16bf af[2], bfr[2][2];
#pragma unroll
        for (int tm = 0; tm < 2; ++tm)
            af[tm] = load_frag(sA(cur), wm * 32 + tm * 16 + r, 16 * h, 16 * h + 8);
#pragma unroll
        for (int mat = 0; mat < NB; ++mat)
#pragma unroll
            for (int tn = 0; tn < 2; ++tn) {
                int row = wn * 32 + tn * 16 + r;
                if constexpr (BF32)
                    bfr[mat][tn] = load_frag_f32((const float*)sB(cur, mat), row, 16 * h);
                else
                    bfr[mat][tn] = load_frag((const __bf16*)sB(cur, mat), row, 16 * h, 16 * h + 8);
            }
#pragma unroll
        for (int tm = 0; tm < 2; ++tm)
#pragma unroll
            for (int tn = 0; tn < 2; ++tn) {
                acc1[tm][tn] = __builtin_amdgcn_wmma_f32_16x16x32_bf16(
                    false, af[tm], false, bfr[0][tn], (short)0, acc1[tm][tn], false, false);
                if constexpr (DUAL)
                    acc2[tm][tn] = __builtin_amdgcn_wmma_f32_16x16x32_bf16(
                        false, af[tm], false, bfr[1][tn], (short)0, acc2[tm][tn], false, false);
            }
    }

    // ---------- rank-320 K-extension ----------
    if constexpr (LR) {
        for (int rb = 0; rb < RP; rb += 32) {
            __syncthreads();
            load_tile_bf16<128>(sA(0), A2_1 + (size_t)m0 * RP + rb, RP, t);
            load_tile_bf16<64>((__bf16*)sB(0, 0), LB1 + (size_t)n0 * RP + rb, RP, t);
            if constexpr (DUAL) {
                load_tile_bf16<128>(sA(1), A2_2 + (size_t)m0 * RP + rb, RP, t);
                load_tile_bf16<64>((__bf16*)sB(0, 1), LB2 + (size_t)n0 * RP + rb, RP, t);
            }
            wait_async<0>();
            __syncthreads();

            v16bf a1[2], a2[2], b1[2], b2[2];
#pragma unroll
            for (int tm = 0; tm < 2; ++tm) {
                a1[tm] = load_frag(sA(0), wm * 32 + tm * 16 + r, 16 * h, 16 * h + 8);
                if constexpr (DUAL)
                    a2[tm] = load_frag(sA(1), wm * 32 + tm * 16 + r, 16 * h, 16 * h + 8);
            }
#pragma unroll
            for (int tn = 0; tn < 2; ++tn) {
                b1[tn] = load_frag((const __bf16*)sB(0, 0), wn * 32 + tn * 16 + r, 16 * h, 16 * h + 8);
                if constexpr (DUAL)
                    b2[tn] = load_frag((const __bf16*)sB(0, 1), wn * 32 + tn * 16 + r, 16 * h, 16 * h + 8);
            }
#pragma unroll
            for (int tm = 0; tm < 2; ++tm)
#pragma unroll
                for (int tn = 0; tn < 2; ++tn) {
                    acc1[tm][tn] = __builtin_amdgcn_wmma_f32_16x16x32_bf16(
                        false, a1[tm], false, b1[tn], (short)0, acc1[tm][tn], false, false);
                    if constexpr (DUAL)
                        acc2[tm][tn] = __builtin_amdgcn_wmma_f32_16x16x32_bf16(
                            false, a2[tm], false, b2[tn], (short)0, acc2[tm][tn], false, false);
                }
        }
    }

    // ---------- epilogue ----------
#pragma unroll
    for (int tm = 0; tm < 2; ++tm)
#pragma unroll
        for (int tn = 0; tn < 2; ++tn)
#pragma unroll
            for (int i = 0; i < 8; ++i) {
                int row = m0 + wm * 32 + tm * 16 + i + 8 * h;
                int col = n0 + wn * 32 + tn * 16 + r;
                size_t idx = (size_t)row * ldc + col;
                if constexpr (MODE == 0) {
                    ((__bf16*)C1)[idx] = f2bf(acc1[tm][tn][i]);
                    ((__bf16*)C2)[idx] = f2bf(acc2[tm][tn][i]);
                } else if constexpr (MODE == 1) {
                    float g = acc1[tm][tn][i];
                    float u = acc2[tm][tn][i];
                    float z = g / (1.0f + __expf(-g)) * u;   // silu(gate)*up
                    ((__bf16*)C1)[idx] = f2bf(z);
                } else if constexpr (MODE == 2) {
                    ((__bf16*)C1)[idx] = f2bf(acc1[tm][tn][i]);
                } else {
                    ((float*)C1)[idx] = acc1[tm][tn][i];
                }
            }
}

// ---------------- host orchestration ----------------
static inline size_t al256(size_t x) { return (x + 255) & ~(size_t)255; }

extern "C" void kernel_launch(void* const* d_in, const int* in_sizes, int n_in,
                              void* d_out, int out_size, void* d_ws, size_t ws_size,
                              hipStream_t stream) {
    (void)in_sizes; (void)n_in; (void)out_size; (void)ws_size;
    const float* x      = (const float*)d_in[0];
    const float* Wmean1 = (const float*)d_in[1];
    const float* Wmean2 = (const float*)d_in[2];
    const float* Wmean3 = (const float*)d_in[3];
    const float* du1    = (const float*)d_in[4];
    const float* dv1    = (const float*)d_in[5];
    const float* du2    = (const float*)d_in[6];
    const float* dv2    = (const float*)d_in[7];
    const float* du3    = (const float*)d_in[8];
    const float* dv3    = (const float*)d_in[9];

    char* ws = (char*)d_ws;
    size_t off = 0;
    auto grab = [&](size_t elems) { __bf16* p = (__bf16*)(ws + off);
                                    off = al256(off + elems * sizeof(__bf16)); return p; };
    __bf16* xb   = grab((size_t)TOK * HID);
    __bf16* zb   = grab((size_t)TOK * INTER);
    __bf16* t1b  = grab((size_t)TOK * RP);
    __bf16* t3b  = grab((size_t)TOK * RP);
    __bf16* t2b  = grab((size_t)TOK * RP);
    __bf16* dvb1 = grab((size_t)RP * HID);
    __bf16* dvb3 = grab((size_t)RP * HID);
    __bf16* dvb2 = grab((size_t)RP * INTER);
    __bf16* dub1 = grab((size_t)INTER * RP);
    __bf16* dub3 = grab((size_t)INTER * RP);
    __bf16* dub2 = grab((size_t)HID * RP);

    auto cvt = [&](const float* s, __bf16* d, int sr, int sc, int dr, int dc) {
        int n = dr * dc;
        cvt_pad_kernel<<<(n + 255) / 256, 256, 0, stream>>>(s, d, sr, sc, dr, dc);
    };
    cvt(x,   xb,   TOK,   HID,   TOK,   HID);
    cvt(dv1, dvb1, RNK,   HID,   RP,    HID);
    cvt(dv3, dvb3, RNK,   HID,   RP,    HID);
    cvt(dv2, dvb2, RNK,   INTER, RP,    INTER);
    cvt(du1, dub1, INTER, RNK,   INTER, RP);
    cvt(du3, dub3, INTER, RNK,   INTER, RP);
    cvt(du2, dub2, HID,   RNK,   HID,   RP);

    dim3 blk(256);
    // stage A: t1 = x@dv1^T, t3 = x@dv3^T  (dual, shares x tiles)
    gemm_ker<0><<<dim3(RP / 64, TOK / 128), blk, 0, stream>>>(
        xb, HID, dvb1, dvb3, HID, nullptr, nullptr, nullptr, nullptr,
        t1b, t3b, RP, HID);
    // stage B: z = silu(x@W1^T + t1@du1^T) * (x@W3^T + t3@du3^T)
    gemm_ker<1><<<dim3(INTER / 64, TOK / 128), blk, 0, stream>>>(
        xb, HID, Wmean1, Wmean3, HID, t1b, t3b, dub1, dub3,
        zb, nullptr, INTER, HID);
    // stage C1: t2 = z@dv2^T
    gemm_ker<2><<<dim3(RP / 64, TOK / 128), blk, 0, stream>>>(
        zb, INTER, dvb2, nullptr, INTER, nullptr, nullptr, nullptr, nullptr,
        t2b, nullptr, RP, INTER);
    // stage C2: out = z@W2^T + t2@du2^T
    gemm_ker<3><<<dim3(HID / 64, TOK / 128), blk, 0, stream>>>(
        zb, INTER, Wmean2, nullptr, INTER, t2b, nullptr, dub2, nullptr,
        d_out, nullptr, HID, INTER);
}
